// Qwen3ForCausalLM_73761768341753
// MI455X (gfx1250) — 16-bit and 8-bit matrix operand order repaired and checked by reading, compiled, not run on hardware
//
#include <hip/hip_runtime.h>
#include <hip/hip_bf16.h>
#include <stdint.h>

#define Dm 2048
#define NHq 16
#define KHv 8
#define HD 128
#define FF 6144
#define NL 2
#define TT 2048
#define VV 32000

typedef __attribute__((ext_vector_type(16))) __bf16 v16bf;
typedef __attribute__((ext_vector_type(8)))  __bf16 v8bf;
typedef __attribute__((ext_vector_type(8)))  float  v8f;
typedef __attribute__((ext_vector_type(16))) int    v16i;
typedef __attribute__((ext_vector_type(4)))  unsigned int tdm_v4u;
typedef __attribute__((ext_vector_type(8)))  int    tdm_v8i;
typedef __attribute__((ext_vector_type(4)))  int    tdm_v4i;

struct __align__(8)  U2 { uint32_t x, y; };
struct __align__(16) U4 { uint32_t x, y, z, w; };

union FragBF { v16bf v; v8bf h[2]; };
union FragA8 { v16i v; U2 d[8]; };
union FragB8 { v16i v; U4 q[4]; };

__device__ __forceinline__ unsigned short f2bf(float f) {
  uint32_t u = __float_as_uint(f);
  uint32_t r = ((u >> 16) & 1u) + 0x7fffu;
  return (unsigned short)((u + r) >> 16);
}

__device__ __forceinline__ unsigned char f2e4m3(float f) {
  uint32_t u = __float_as_uint(f);
  uint32_t s = (u >> 24) & 0x80u;
  float a = __builtin_fabsf(f);
  if (a >= 448.f) return (unsigned char)(s | 0x7e);
  uint32_t b = __float_as_uint(a);
  int e = (int)((b >> 23) & 0xffu) - 127;
  uint32_t m3 = ((b & 0x7fffffu) + 0x80000u) >> 20;
  if (m3 == 8u) { m3 = 0u; e++; }
  if (e < -6) {
    int sh = -6 - e;
    if (sh > 3) return (unsigned char)s;
    return (unsigned char)(s | ((8u | m3) >> sh));
  }
  return (unsigned char)(s | ((uint32_t)(e + 7) << 3) | m3);
}

__device__ __forceinline__ v8f wmma_bf16(v16bf a, v16bf b, v8f c) {
  return __builtin_amdgcn_wmma_f32_16x16x32_bf16(false, a, false, b, (short)0, c,
                                                 false, false);
}
__device__ __forceinline__ v8f wmma_fp8(v16i a, v16i b, v8f c) {
  return __builtin_amdgcn_wmma_f32_16x16x128_fp8_fp8(a, b, (short)0, c, false, false);
}

__device__ __forceinline__ void async_copy16(void* lds_dst, const void* gsrc) {
  uint32_t l = (uint32_t)(size_t)lds_dst;
  asm volatile("global_load_async_to_lds_b128 %0, %1, off"
               :: "v"(l), "v"(gsrc) : "memory");
}
__device__ __forceinline__ void async_wait0() {
  asm volatile("s_wait_asynccnt 0" ::: "memory");
}

__device__ __forceinline__ void tdm_load_2d(uint32_t lds_addr, const void* gaddr,
                                            uint32_t row_bytes, uint32_t rows,
                                            uint32_t tensor_w, uint32_t tensor_h,
                                            uint64_t stride0) {
  const uint64_t ga = (uint64_t)(size_t)gaddr;
  tdm_v4u g0 = { 1u,
                 lds_addr,
                 (uint32_t)ga,
                 (uint32_t)((ga >> 32) & 0x1ffffffu) | (2u << 30) };
  const uint32_t w0 = (1u << 20) | (4u << 22) | (3u << 25);
  tdm_v8i g1 = { (int)w0,
                 (int)((tensor_w & 0xffffu) << 16),
                 (int)(((tensor_w >> 16) & 0xffffu) | ((tensor_h & 0xffffu) << 16)),
                 (int)(((tensor_h >> 16) & 0xffffu) | ((row_bytes & 0xffffu) << 16)),
                 (int)(rows & 0xffffu),
                 (int)(uint32_t)stride0,
                 (int)(uint32_t)((stride0 >> 32) & 0xffffu),
                 0 };
  tdm_v4i z4 = { 0, 0, 0, 0 };
  tdm_v8i z8 = { 0, 0, 0, 0, 0, 0, 0, 0 };
  __builtin_amdgcn_tensor_load_to_lds(g0, g1, z4, z4, z8, 0);
}

__global__ void k_gather(const int* __restrict__ ids, const float* __restrict__ E,
                         float* __restrict__ H) {
  const int t = blockIdx.x;
  const size_t src = (size_t)ids[t] * Dm;
  const size_t dst = (size_t)t * Dm;
  for (int i = threadIdx.x; i < Dm; i += blockDim.x) H[dst + i] = E[src + i];
}

__global__ __launch_bounds__(256) void k_rmsnorm(const float* __restrict__ X,
                                                 const float* __restrict__ W,
                                                 float* __restrict__ Y, int width) {
  __shared__ float red[8];
  const size_t row = (size_t)blockIdx.x * width;
  float p = 0.f;
  for (int i = threadIdx.x; i < width; i += 256) { float v = X[row + i]; p += v * v; }
  for (int off = 16; off > 0; off >>= 1) p += __shfl_xor(p, off, 32);
  const int lane = threadIdx.x & 31, wv = threadIdx.x >> 5;
  if (lane == 0) red[wv] = p;
  __syncthreads();
  if (wv == 0) {
    float sv = (lane < 8) ? red[lane] : 0.f;
    for (int off = 4; off > 0; off >>= 1) sv += __shfl_xor(sv, off, 8);
    if (lane == 0) red[0] = sv;
  }
  __syncthreads();
  const float rs = rsqrtf(red[0] / (float)width + 1e-6f);
  for (int i = threadIdx.x; i < width; i += 256) Y[row + i] = X[row + i] * rs * W[i];
}

__global__ void k_rope_bf16(const float* __restrict__ X, unsigned short* __restrict__ Y,
                            const int* __restrict__ pos, int nH) {
  const int t = blockIdx.x, hh = blockIdx.y, j = threadIdx.x;
  const size_t base = ((size_t)t * nH + hh) * HD;
  const float inv = __powf(1.0e6f, -(float)j / 64.f);
  const float fr = (float)pos[t] * inv;
  float sn, cs; __sincosf(fr, &sn, &cs);
  const float x1 = X[base + j], x2 = X[base + j + 64];
  Y[base + j]      = f2bf(x1 * cs - x2 * sn);
  Y[base + j + 64] = f2bf(x2 * cs + x1 * sn);
}

__global__ void k_cvt_bf16(const float* __restrict__ X, unsigned short* __restrict__ Y,
                           size_t n) {
  size_t i = (size_t)blockIdx.x * 256 + threadIdx.x;
  if (i < n) Y[i] = f2bf(X[i]);
}

__global__ void k_cvt_fp8(const float* __restrict__ X, unsigned char* __restrict__ Y,
                          size_t n) {
  size_t i = (size_t)blockIdx.x * 256 + threadIdx.x;
  if (i < n) Y[i] = f2e4m3(X[i]);
}

__global__ void k_silu_mul(float* __restrict__ G, const float* __restrict__ U, size_t n) {
  size_t i = (size_t)blockIdx.x * 256 + threadIdx.x;
  if (i < n) { float x = G[i]; G[i] = (x / (1.f + __expf(-x))) * U[i]; }
}

__global__ __launch_bounds__(256) void k_wt_bf16(const float* __restrict__ W,
                                                 unsigned short* __restrict__ Wt,
                                                 int Kd, int Nn) {
  __shared__ float tile[32][33];
  const int kb = blockIdx.y * 32, nb = blockIdx.x * 32;
  const int tx = threadIdx.x & 31, ty = threadIdx.x >> 5;
  for (int r = 0; r < 32; r += 8)
    tile[ty + r][tx] = W[(size_t)(kb + ty + r) * Nn + nb + tx];
  __syncthreads();
  for (int r = 0; r < 32; r += 8)
    Wt[(size_t)(nb + ty + r) * Kd + kb + tx] = f2bf(tile[tx][ty + r]);
}

__global__ __launch_bounds__(256) void k_wt_fp8(const float* __restrict__ W,
                                                unsigned char* __restrict__ Wt,
                                                int Kd, int Nn) {
  __shared__ float tile[32][33];
  const int kb = blockIdx.y * 32, nb = blockIdx.x * 32;
  const int tx = threadIdx.x & 31, ty = threadIdx.x >> 5;
  for (int r = 0; r < 32; r += 8)
    tile[ty + r][tx] = W[(size_t)(kb + ty + r) * Nn + nb + tx];
  __syncthreads();
  for (int r = 0; r < 32; r += 8)
    Wt[(size_t)(nb + ty + r) * Kd + kb + tx] = f2e4m3(tile[tx][ty + r]);
}

__global__ __launch_bounds__(256) void k_gemm_bf16(
    const unsigned short* __restrict__ A, const unsigned short* __restrict__ Bt,
    float* __restrict__ C, int M, int Nn, int Kd,
    const float* __restrict__ colScale, const float* __restrict__ residual) {
  __shared__ __align__(16) unsigned short As[128][40];
  __shared__ __align__(16) unsigned short Bs[128][40];
  const int tid = threadIdx.x;
  const int lane = tid & 31, wave = tid >> 5;
  const int hi = lane >> 4, l15 = lane & 15;
  const int m0 = blockIdx.y * 128, n0 = blockIdx.x * 128;
  const int wr = wave >> 2, wc = wave & 3;

  v8f acc[4][2];
  for (int i = 0; i < 4; i++)
    for (int j = 0; j < 2; j++)
      for (int e = 0; e < 8; e++) acc[i][j][e] = 0.f;

  for (int k0 = 0; k0 < Kd; k0 += 32) {
    for (int i = tid; i < 512; i += 256) {
      const int r = i >> 2, c = i & 3;
      async_copy16(&As[r][c * 8], A + (size_t)(m0 + r) * Kd + k0 + c * 8);
    }
    for (int i = tid; i < 512; i += 256) {
      const int n = i >> 2, c = i & 3;
      async_copy16(&Bs[n][c * 8], Bt + (size_t)(n0 + n) * Kd + k0 + c * 8);
    }
    if (k0 + 32 < Kd)
      __builtin_prefetch(A + (size_t)(m0 + (tid >> 1)) * Kd + k0 + 32, 0, 0);
    async_wait0();
    __syncthreads();

    FragBF af[4], bfr[2];
    for (int i = 0; i < 4; i++) {
      const int m = wr * 64 + i * 16 + l15;
      af[i].h[0] = *(const v8bf*)&As[m][hi * 8];
      af[i].h[1] = *(const v8bf*)&As[m][16 + hi * 8];
    }
    for (int j = 0; j < 2; j++) {
      const int n = wc * 32 + j * 16 + l15;
      bfr[j].h[0] = *(const v8bf*)&Bs[n][hi * 8];
      bfr[j].h[1] = *(const v8bf*)&Bs[n][16 + hi * 8];
    }
    for (int i = 0; i < 4; i++)
      for (int j = 0; j < 2; j++) acc[i][j] = wmma_bf16(af[i].v, bfr[j].v, acc[i][j]);
    __syncthreads();
  }

  for (int i = 0; i < 4; i++)
    for (int j = 0; j < 2; j++) {
      const int rowb = m0 + wr * 64 + i * 16 + hi * 8;
      const int col = n0 + wc * 32 + j * 16 + l15;
      const float cs = colScale ? colScale[col] : 1.f;
      for (int r = 0; r < 8; r++) {
        const size_t idx = (size_t)(rowb + r) * Nn + col;
        float v = acc[i][j][r] * cs;
        if (residual) v += residual[idx];
        C[idx] = v;
      }
    }
}

__global__ __launch_bounds__(256) void k_gemm_fp8(
    const unsigned char* __restrict__ A8, const unsigned char* __restrict__ B8t,
    float* __restrict__ C, int M, int Nn, int Kd,
    const float* __restrict__ colScale, const float* __restrict__ residual) {
  __shared__ __align__(16) unsigned char As8[128][144];
  __shared__ __align__(16) unsigned char Bs8[128][144];
  const int tid = threadIdx.x;
  const int lane = tid & 31, wave = tid >> 5;
  const int hi = lane >> 4, l15 = lane & 15;
  const int m0 = blockIdx.y * 128, n0 = blockIdx.x * 128;
  const int wr = wave >> 2, wc = wave & 3;

  v8f acc[4][2];
  for (int i = 0; i < 4; i++)
    for (int j = 0; j < 2; j++)
      for (int e = 0; e < 8; e++) acc[i][j][e] = 0.f;

  for (int k0 = 0; k0 < Kd; k0 += 128) {
    if (wave == 0) {
      tdm_load_2d((uint32_t)(size_t)&As8[0][0], A8 + (size_t)m0 * Kd + k0,
                  128, 128, (uint32_t)Kd, (uint32_t)M, (uint64_t)Kd);
      tdm_load_2d((uint32_t)(size_t)&Bs8[0][0], B8t + (size_t)n0 * Kd + k0,
                  128, 128, (uint32_t)Kd, (uint32_t)Nn, (uint64_t)Kd);
      __builtin_amdgcn_s_wait_tensorcnt(0);
    }
    __syncthreads();

    FragA8 a8[4]; FragB8 b8[2];
    for (int i = 0; i < 4; i++) {
      const int m = wr * 64 + i * 16 + l15;
      for (int c = 0; c < 8; c++) {
        a8[i].d[c]     = *(const U2*)&As8[m][32 * (c >> 1) + hi * 16 + 8 * (c & 1)];
      }
    }
    for (int j = 0; j < 2; j++) {
      const int n = wc * 32 + j * 16 + l15;
      for (int g = 0; g < 4; g++)
        b8[j].q[g] = *(const U4*)&Bs8[n][32 * g + hi * 16];
    }
    for (int i = 0; i < 4; i++)
      for (int j = 0; j < 2; j++) acc[i][j] = wmma_fp8(a8[i].v, b8[j].v, acc[i][j]);
    __syncthreads();
  }

  for (int i = 0; i < 4; i++)
    for (int j = 0; j < 2; j++) {
      const int rowb = m0 + wr * 64 + i * 16 + hi * 8;
      const int col = n0 + wc * 32 + j * 16 + l15;
      const float cs = colScale ? colScale[col] : 1.f;
      for (int r = 0; r < 8; r++) {
        const size_t idx = (size_t)(rowb + r) * Nn + col;
        float v = acc[i][j][r] * cs;
        if (residual) v += residual[idx];
        C[idx] = v;
      }
    }
}

__global__ __launch_bounds__(128) void k_flash(
    const unsigned short* __restrict__ Qb, const unsigned short* __restrict__ Kb,
    const unsigned short* __restrict__ Vb, float* __restrict__ O) {
  __shared__ __align__(16) unsigned short Vs[128][72];
  __shared__ __align__(16) unsigned short Ps[4][16][72];
  const int lane = threadIdx.x & 31, wave = threadIdx.x >> 5;
  const int hi = lane >> 4, l15 = lane & 15;
  const int qblk = blockIdx.x, head = blockIdx.y;
  const int kvh = head / (NHq / KHv);
  const int qrow0 = qblk * 64 + wave * 16;
  const float sm = 0.088388347648318447f;

  FragBF qf[4];
  {
    const int m = qrow0 + l15;
    const unsigned short* qrow = Qb + ((size_t)m * NHq + head) * HD;
    for (int c = 0; c < 4; c++) {
      qf[c].h[0] = *(const v8bf*)&qrow[32 * c + hi * 8];
      qf[c].h[1] = *(const v8bf*)&qrow[32 * c + 16 + hi * 8];
    }
  }

  float mrun[8], lrun[8];
  v8f oacc[8];
  for (int r = 0; r < 8; r++) { mrun[r] = -1e30f; lrun[r] = 0.f; }
  for (int ht = 0; ht < 8; ht++)
    for (int e = 0; e < 8; e++) oacc[ht][e] = 0.f;

  for (int kb = 0; kb <= qblk; kb++) {
    const int t0 = kb * 64;
    for (int i = threadIdx.x; i < 64 * 128; i += 128) {
      const int h = i & 127, kv = i >> 7;
      Vs[h][kv] = Vb[((size_t)(t0 + kv) * KHv + kvh) * HD + h];
    }
    __syncthreads();

    v8f sc[4];
    for (int ct = 0; ct < 4; ct++) {
      for (int e = 0; e < 8; e++) sc[ct][e] = 0.f;
      const int kvr = t0 + ct * 16 + l15;
      const unsigned short* krow = Kb + ((size_t)kvr * KHv + kvh) * HD;
      for (int c = 0; c < 4; c++) {
        FragBF kf;
        kf.h[0] = *(const v8bf*)&krow[32 * c + hi * 8];
        kf.h[1] = *(const v8bf*)&krow[32 * c + 16 + hi * 8];
        sc[ct] = wmma_bf16(qf[c].v, kf.v, sc[ct]);
      }
    }

    float alpha[8];
    for (int r = 0; r < 8; r++) {
      const int qr = qrow0 + r + hi * 8;
      float mx = -1e30f;
      for (int ct = 0; ct < 4; ct++) {
        const int col = t0 + ct * 16 + l15;
        float s = sc[ct][r] * sm;
        if (col > qr) s = -1e30f;
        sc[ct][r] = s;
        mx = fmaxf(mx, s);
      }
      for (int off = 8; off > 0; off >>= 1) mx = fmaxf(mx, __shfl_xor(mx, off, 16));
      const float mn = fmaxf(mrun[r], mx);
      alpha[r] = __expf(mrun[r] - mn);
      mrun[r] = mn;
    }
    for (int r = 0; r < 8; r++) {
      float rs = 0.f;
      const int mrow = r + hi * 8;
      for (int ct = 0; ct < 4; ct++) {
        const float p = __expf(sc[ct][r] - mrun[r]);
        rs += p;
        Ps[wave][mrow][ct * 16 + l15] = f2bf(p);
      }
      for (int off = 8; off > 0; off >>= 1) rs += __shfl_xor(rs, off, 16);
      lrun[r] = lrun[r] * alpha[r] + rs;
    }
    for (int ht = 0; ht < 8; ht++)
      for (int r = 0; r < 8; r++) oacc[ht][r] *= alpha[r];

    asm volatile("s_wait_dscnt 0" ::: "memory");

    FragBF pf[2];
    for (int c = 0; c < 2; c++) {
      pf[c].h[0] = *(const v8bf*)&Ps[wave][l15][32 * c + hi * 8];
      pf[c].h[1] = *(const v8bf*)&Ps[wave][l15][32 * c + 16 + hi * 8];
    }
    for (int ht = 0; ht < 8; ht++) {
      const int h = ht * 16 + l15;
      for (int c = 0; c < 2; c++) {
        FragBF vf;
        vf.h[0] = *(const v8bf*)&Vs[h][32 * c + hi * 8];
        vf.h[1] = *(const v8bf*)&Vs[h][32 * c + 16 + hi * 8];
        oacc[ht] = wmma_bf16(pf[c].v, vf.v, oacc[ht]);
      }
    }
    __syncthreads();
  }

  for (int ht = 0; ht < 8; ht++)
    for (int r = 0; r < 8; r++) {
      const int row = qrow0 + r + hi * 8;
      const int h = ht * 16 + l15;
      O[((size_t)row * NHq + head) * HD + h] = oacc[ht][r] / lrun[r];
    }
}

extern "C" void kernel_launch(void* const* d_in, const int* in_sizes, int n_in,
                              void* d_out, int out_size, void* d_ws, size_t ws_size,
                              hipStream_t stream) {
  const int*   ids       = (const int*)d_in[0];
  const int*   pos       = (const int*)d_in[1];
  const float* embed     = (const float*)d_in[2];
  const float* ln1       = (const float*)d_in[3];
  const float* q_w       = (const float*)d_in[4];
  const float* q_norm    = (const float*)d_in[5];
  const float* k_w       = (const float*)d_in[6];
  const float* k_norm    = (const float*)d_in[7];
  const float* v_w       = (const float*)d_in[8];
  const float* o_w       = (const float*)d_in[9];
  const float* ln2       = (const float*)d_in[10];
  const float* gate_w    = (const float*)d_in[11];
  const float* gate_sc   = (const float*)d_in[12];
  const float* up_w      = (const float*)d_in[13];
  const float* up_sc     = (const float*)d_in[14];
  const float* down_w    = (const float*)d_in[15];
  const float* down_sc   = (const float*)d_in[16];
  const float* fin_norm  = (const float*)d_in[17];
  const float* lm_head   = (const float*)d_in[18];
  float* out = (float*)d_out;
  (void)in_sizes; (void)n_in; (void)out_size; (void)ws_size;

  char* base = (char*)d_ws;
  size_t off = 0;
  auto carve = [&](size_t bytes) -> void* {
    void* p = base + off;
    off += (bytes + 255) & ~(size_t)255;
    return p;
  };
  float* h  = (float*)carve((size_t)TT * Dm * 4);
  float* xn = (float*)carve((size_t)TT * Dm * 4);
  float* qf = (float*)carve((size_t)TT * NHq * HD * 4);
  float* kf = (float*)carve((size_t)TT * KHv * HD * 4);
  float* vf = (float*)carve((size_t)TT * KHv * HD * 4);
  float* ao = (float*)carve((size_t)TT * NHq * HD * 4);
  float* mg = (float*)carve((size_t)TT * FF * 4);
  float* mu = (float*)carve((size_t)TT * FF * 4);
  unsigned short* qB = (unsigned short*)carve((size_t)TT * NHq * HD * 2);
  unsigned short* kB = (unsigned short*)carve((size_t)TT * KHv * HD * 2);
  unsigned short* vB = (unsigned short*)carve((size_t)TT * KHv * HD * 2);
  unsigned short* xb = (unsigned short*)carve((size_t)TT * Dm * 2);
  unsigned char*  xq = (unsigned char*)carve((size_t)TT * FF);
  unsigned short* wb = (unsigned short*)carve((size_t)VV * Dm * 2);
  unsigned char*  w8 = (unsigned char*)carve((size_t)Dm * FF);

  const size_t TD = (size_t)TT * Dm;

  k_gather<<<TT, 256, 0, stream>>>(ids, embed, h);

  for (int l = 0; l < NL; l++) {
    const float* qw = q_w    + (size_t)l * Dm * (NHq * HD);
    const float* kw = k_w    + (size_t)l * Dm * (KHv * HD);
    const float* vw = v_w    + (size_t)l * Dm * (KHv * HD);
    const float* ow = o_w    + (size_t)l * (NHq * HD) * Dm;
    const float* gw = gate_w + (size_t)l * Dm * FF;
    const float* uw = up_w   + (size_t)l * Dm * FF;
    const float* dw = down_w + (size_t)l * FF * Dm;

    k_rmsnorm<<<TT, 256, 0, stream>>>(h, ln1 + (size_t)l * Dm, xn, Dm);
    k_cvt_bf16<<<(TD + 255) / 256, 256, 0, stream>>>(xn, xb, TD);
    k_wt_bf16<<<dim3((NHq * HD) / 32, Dm / 32), 256, 0, stream>>>(qw, wb, Dm, NHq * HD);
    k_gemm_bf16<<<dim3((NHq * HD) / 128, TT / 128), 256, 0, stream>>>(
        xb, wb, qf, TT, NHq * HD, Dm, nullptr, nullptr);
    k_wt_bf16<<<dim3((KHv * HD) / 32, Dm / 32), 256, 0, stream>>>(kw, wb, Dm, KHv * HD);
    k_gemm_bf16<<<dim3((KHv * HD) / 128, TT / 128), 256, 0, stream>>>(
        xb, wb, kf, TT, KHv * HD, Dm, nullptr, nullptr);
    k_wt_bf16<<<dim3((KHv * HD) / 32, Dm / 32), 256, 0, stream>>>(vw, wb, Dm, KHv * HD);
    k_gemm_bf16<<<dim3((KHv * HD) / 128, TT / 128), 256, 0, stream>>>(
        xb, wb, vf, TT, KHv * HD, Dm, nullptr, nullptr);
    k_rmsnorm<<<TT * NHq, 256, 0, stream>>>(qf, q_norm + (size_t)l * HD, qf, HD);
    k_rmsnorm<<<TT * KHv, 256, 0, stream>>>(kf, k_norm + (size_t)l * HD, kf, HD);
    k_rope_bf16<<<dim3(TT, NHq), 64, 0, stream>>>(qf, qB, pos, NHq);
    k_rope_bf16<<<dim3(TT, KHv), 64, 0, stream>>>(kf, kB, pos, KHv);
    k_cvt_bf16<<<((size_t)TT * KHv * HD + 255) / 256, 256, 0, stream>>>(
        vf, vB, (size_t)TT * KHv * HD);
    k_flash<<<dim3(TT / 64, NHq), 128, 0, stream>>>(qB, kB, vB, ao);
    k_cvt_bf16<<<(TD + 255) / 256, 256, 0, stream>>>(ao, xb, TD);
    k_wt_bf16<<<dim3(Dm / 32, (NHq * HD) / 32), 256, 0, stream>>>(ow, wb, NHq * HD, Dm);
    k_gemm_bf16<<<dim3(Dm / 128, TT / 128), 256, 0, stream>>>(
        xb, wb, h, TT, Dm, NHq * HD, nullptr, h);

    k_rmsnorm<<<TT, 256, 0, stream>>>(h, ln2 + (size_t)l * Dm, xn, Dm);
    k_cvt_fp8<<<(TD + 255) / 256, 256, 0, stream>>>(xn, xq, TD);
    k_wt_fp8<<<dim3(FF / 32, Dm / 32), 256, 0, stream>>>(gw, w8, Dm, FF);
    k_gemm_fp8<<<dim3(FF / 128, TT / 128), 256, 0, stream>>>(
        xq, w8, mg, TT, FF, Dm, gate_sc + (size_t)l * FF, nullptr);
    k_wt_fp8<<<dim3(FF / 32, Dm / 32), 256, 0, stream>>>(uw, w8, Dm, FF);
    k_gemm_fp8<<<dim3(FF / 128, TT / 128), 256, 0, stream>>>(
        xq, w8, mu, TT, FF, Dm, up_sc + (size_t)l * FF, nullptr);
    k_silu_mul<<<((size_t)TT * FF + 255) / 256, 256, 0, stream>>>(
        mg, mu, (size_t)TT * FF);
    k_cvt_fp8<<<((size_t)TT * FF + 255) / 256, 256, 0, stream>>>(
        mg, xq, (size_t)TT * FF);
    k_wt_fp8<<<dim3(Dm / 32, FF / 32), 256, 0, stream>>>(dw, w8, FF, Dm);
    k_gemm_fp8<<<dim3(Dm / 128, TT / 128), 256, 0, stream>>>(
        xq, w8, h, TT, Dm, FF, down_sc + (size_t)l * Dm, h);
  }

  k_rmsnorm<<<TT, 256, 0, stream>>>(h, fin_norm, xn, Dm);
  k_cvt_bf16<<<(TD + 255) / 256, 256, 0, stream>>>(xn, xb, TD);
  k_wt_bf16<<<dim3(VV / 32, Dm / 32), 256, 0, stream>>>(lm_head, wb, Dm, VV);
  k_gemm_bf16<<<dim3(VV / 128, TT / 128), 256, 0, stream>>>(
      xb, wb, out, TT, VV, Dm, nullptr, nullptr);
}
